// MLPStateEncoder_80556406604087
// MI455X (gfx1250) — matrix operand order repaired and checked by reading, not compiled, not run on hardware
//
#include <hip/hip_runtime.h>
#include <hip/hip_bf16.h>
#include <cstdint>

#define HDIM   256
#define NNODE  32
#define NBATCH 64
#define NEDGE  992
#define NIN    8
#define NOUT   4
#define BN_EPS 1e-5f

typedef __attribute__((ext_vector_type(16))) __bf16 v16bf;
typedef __attribute__((ext_vector_type(8)))  __bf16 v8bf;
typedef __attribute__((ext_vector_type(8)))  float  v8f;

__device__ __forceinline__ __bf16 f2bf(float x) { return (__bf16)x; }
__device__ __forceinline__ float elu1(float x) { return x > 0.f ? x : (__expf(x) - 1.f); }

// ---- extract one-hot indices from rel matrices -----------------------------
__global__ void k_extract_idx(const float* __restrict__ rel, int* __restrict__ idx) {
    int e = blockIdx.x * blockDim.x + threadIdx.x;
    if (e >= NEDGE) return;
    int best = 0; float bv = rel[e * NNODE];
    for (int n = 1; n < NNODE; ++n) {
        float v = rel[e * NNODE + n];
        if (v > bv) { bv = v; best = n; }
    }
    idx[e] = best;
}

// ---- pack f32 weight [K,HDIM] into per-lane WMMA B fragments (bf16) --------
__global__ void k_pack_w(const float* __restrict__ W, __bf16* __restrict__ out, int K) {
    int Kt = K >> 5;
    int total = Kt * (HDIM >> 4) * 32;
    int tid = blockIdx.x * blockDim.x + threadIdx.x;
    if (tid >= total) return;
    int lane = tid & 31;
    int t    = tid >> 5;
    int nt   = t / Kt, kt = t - nt * Kt;
    int n    = nt * 16 + (lane & 15);
    int kb   = kt * 32 + (lane >> 4) * 8;
    __bf16* dst = out + (size_t)t * 512 + lane * 16;
#pragma unroll
    for (int j = 0; j < 16; ++j)
        dst[j] = f2bf(W[(size_t)(kb + (j & 7) + ((j >> 3) << 4)) * HDIM + n]);
}

// ---- small K=8 first layer of mlp1 -> bf16 ---------------------------------
__global__ void k_fc8(const float* __restrict__ X, const float* __restrict__ W,
                      const float* __restrict__ bias, __bf16* __restrict__ Y, int M) {
    int i = blockIdx.x * blockDim.x + threadIdx.x;
    if (i >= M * HDIM) return;
    int r = i >> 8, c = i & 255;
    float acc = bias[c];
#pragma unroll
    for (int k = 0; k < NIN; ++k) acc += X[r * NIN + k] * W[k * HDIM + c];
    Y[i] = f2bf(elu1(acc));
}

// ---- WMMA bf16 GEMM: out = ELU(A[M,K] * W + bias), K = numSeg*256 ----------
// A is bf16. mode 0: plain (lda=K). mode 1: gather [recv|send]. mode 2: +skip.
// Writes bf16 to Cb if non-null, else f32 to Cf.
__global__ __launch_bounds__(256)
void k_gemm_bf16(const __bf16* __restrict__ A, const __bf16* __restrict__ nodeX,
                 const __bf16* __restrict__ skipX,
                 const int* __restrict__ recI, const int* __restrict__ sendI,
                 const __bf16* __restrict__ PB, const float* __restrict__ bias,
                 float* __restrict__ Cf, __bf16* __restrict__ Cb,
                 int M, int numSeg, int mode) {
    const int lane = threadIdx.x & 31;
    const int wid  = threadIdx.x >> 5;
    const int wr = wid & 1, wc = wid >> 1;          // 2x4 wave grid
    const int rowBase = blockIdx.x * 128 + wr * 64;
    const int colBase = wc * 64;
    const int Kt = numSeg * 8;
    const int K  = numSeg * 256;
    const int lh8 = (lane >> 4) * 8;                // per-lane K sub-offset

    // Per-segment, per-row-fragment base pointers (lh8 pre-folded). All
    // indexing below is compile-time constant so these stay in registers.
    const __bf16* segA[3][4];
#pragma unroll
    for (int a = 0; a < 4; ++a) {
        int r = rowBase + a * 16 + (lane & 15);
        if (r >= M) r = M - 1;
        if (mode == 0) {
#pragma unroll
            for (int s = 0; s < 3; ++s) segA[s][a] = A + (size_t)r * K + s * 256 + lh8;
        } else {
            int b = r / NEDGE, e = r - b * NEDGE;
            segA[0][a] = nodeX + ((size_t)b * NNODE + recI[e])  * HDIM + lh8;
            segA[1][a] = nodeX + ((size_t)b * NNODE + sendI[e]) * HDIM + lh8;
            segA[2][a] = (mode == 2) ? (skipX + (size_t)r * HDIM + lh8) : segA[0][a];
        }
    }
    const __bf16* pbB[4];
#pragma unroll
    for (int b = 0; b < 4; ++b)
        pbB[b] = PB + (size_t)((wc * 4 + b) * Kt) * 512 + lane * 16;

    v8f acc[4][4];
#pragma unroll
    for (int a = 0; a < 4; ++a)
#pragma unroll
        for (int b = 0; b < 4; ++b) acc[a][b] = (v8f)(0.f);

    auto do_seg = [&](int s) {   // s is always a literal at call sites
#pragma unroll
        for (int k8 = 0; k8 < 8; ++k8) {
            v16bf af[4];
#pragma unroll
            for (int a = 0; a < 4; ++a) {
                const v8bf* p = (const v8bf*)(segA[s][a] + (k8 << 5));
                af[a] = __builtin_shufflevector(p[0], p[2], 0, 1, 2, 3, 4, 5, 6, 7,
                                                8, 9, 10, 11, 12, 13, 14, 15);
            }
            v16bf bfr[4];
#pragma unroll
            for (int b = 0; b < 4; ++b)
                bfr[b] = *(const v16bf*)(pbB[b] + (s * 8 + k8) * 512);
#pragma unroll
            for (int a = 0; a < 4; ++a)
#pragma unroll
                for (int b = 0; b < 4; ++b)
                    acc[a][b] = __builtin_amdgcn_wmma_f32_16x16x32_bf16(
                        false, af[a], false, bfr[b], (short)0, acc[a][b], false, false);
        }
    };
    do_seg(0);
    if (numSeg > 1) do_seg(1);
    if (numSeg > 2) do_seg(2);

    const int col16 = lane & 15;
    const int rsh   = (lane >> 4) * 8;
    if (Cb) {
#pragma unroll
        for (int b = 0; b < 4; ++b) {
            int gCol = colBase + b * 16 + col16;
            float bv = bias[gCol];
#pragma unroll
            for (int a = 0; a < 4; ++a)
#pragma unroll
                for (int v = 0; v < 8; ++v) {
                    int gRow = rowBase + a * 16 + rsh + v;
                    if (gRow < M)
                        Cb[(size_t)gRow * HDIM + gCol] = f2bf(elu1(acc[a][b][v] + bv));
                }
        }
    } else {
#pragma unroll
        for (int b = 0; b < 4; ++b) {
            int gCol = colBase + b * 16 + col16;
            float bv = bias[gCol];
#pragma unroll
            for (int a = 0; a < 4; ++a)
#pragma unroll
                for (int v = 0; v < 8; ++v) {
                    int gRow = rowBase + a * 16 + rsh + v;
                    if (gRow < M)
                        Cf[(size_t)gRow * HDIM + gCol] = elu1(acc[a][b][v] + bv);
                }
        }
    }
}

// ---- batchnorm: deterministic two-stage column reduction -------------------
__global__ void k_bn_partial(const float* __restrict__ X, float* __restrict__ ps,
                             float* __restrict__ ps2, int M, int chunkRows) {
    int c = threadIdx.x;
    int r0 = blockIdx.x * chunkRows;
    int r1 = r0 + chunkRows; if (r1 > M) r1 = M;
    float s = 0.f, s2 = 0.f;
    for (int r = r0; r < r1; ++r) {
        float v = X[(size_t)r * HDIM + c];
        s += v; s2 += v * v;
    }
    ps [blockIdx.x * HDIM + c] = s;
    ps2[blockIdx.x * HDIM + c] = s2;
}

__global__ void k_bn_final(const float* __restrict__ ps, const float* __restrict__ ps2,
                           float* __restrict__ mean, float* __restrict__ istd,
                           int nChunks, int M) {
    int c = threadIdx.x;
    float s = 0.f, s2 = 0.f;
    for (int k = 0; k < nChunks; ++k) { s += ps[k * HDIM + c]; s2 += ps2[k * HDIM + c]; }
    float m = s / (float)M;
    float v = s2 / (float)M - m * m;
    mean[c] = m;
    istd[c] = rsqrtf(v + BN_EPS);
}

// BN apply in f32 (in place); optionally also emit bf16 copy for next GEMM.
__global__ void k_bn_apply(float* __restrict__ X, const float* __restrict__ mean,
                           const float* __restrict__ istd, const float* __restrict__ g,
                           const float* __restrict__ b, __bf16* __restrict__ Xb, int M) {
    int i = blockIdx.x * blockDim.x + threadIdx.x;
    if (i >= M * HDIM) return;
    int c = i & 255;
    float y = (X[i] - mean[c]) * istd[c] * g[c] + b[c];
    X[i] = y;
    if (Xb) Xb[i] = f2bf(y);
}

// ---- edge2node: per (b,n) sum edges whose receiver == n, / NNODE -> bf16 ---
__global__ void k_edge2node(const float* __restrict__ Xe, const int* __restrict__ recI,
                            __bf16* __restrict__ Xn) {
    __shared__ int srec[NEDGE];
    for (int e = threadIdx.x; e < NEDGE; e += blockDim.x) srec[e] = recI[e];
    __syncthreads();
    int bn = blockIdx.x;
    int b = bn / NNODE, n = bn - b * NNODE;
    int c = threadIdx.x;
    float s = 0.f;
    for (int e = 0; e < NEDGE; ++e)
        if (srec[e] == n) s += Xe[((size_t)b * NEDGE + e) * HDIM + c];
    Xn[(size_t)bn * HDIM + c] = f2bf(s * (1.f / NNODE));
}

// ---- final 256->4 projection + softmax (one wave per row) ------------------
__global__ void k_out_softmax(const float* __restrict__ X, const float* __restrict__ W,
                              const float* __restrict__ bo, float* __restrict__ out, int M) {
    int lane = threadIdx.x & 31;
    int row  = blockIdx.x * 8 + (threadIdx.x >> 5);
    if (row >= M) return;
    const float* x = X + (size_t)row * HDIM;
    float a0 = 0, a1 = 0, a2 = 0, a3 = 0;
    for (int k = lane; k < HDIM; k += 32) {
        float v = x[k];
        const float* w = W + k * 4;
        a0 += v * w[0]; a1 += v * w[1]; a2 += v * w[2]; a3 += v * w[3];
    }
#pragma unroll
    for (int off = 16; off; off >>= 1) {
        a0 += __shfl_xor(a0, off, 32);
        a1 += __shfl_xor(a1, off, 32);
        a2 += __shfl_xor(a2, off, 32);
        a3 += __shfl_xor(a3, off, 32);
    }
    if (lane == 0) {
        a0 += bo[0]; a1 += bo[1]; a2 += bo[2]; a3 += bo[3];
        float mx = fmaxf(fmaxf(a0, a1), fmaxf(a2, a3));
        float e0 = __expf(a0 - mx), e1 = __expf(a1 - mx), e2 = __expf(a2 - mx), e3 = __expf(a3 - mx);
        float inv = 1.f / (e0 + e1 + e2 + e3);
        float* o = out + (size_t)row * 4;
        o[0] = e0 * inv; o[1] = e1 * inv; o[2] = e2 * inv; o[3] = e3 * inv;
    }
}

// ---------------------------------------------------------------------------
extern "C" void kernel_launch(void* const* d_in, const int* in_sizes, int n_in,
                              void* d_out, int out_size, void* d_ws, size_t ws_size,
                              hipStream_t stream) {
    (void)out_size; (void)ws_size;
    const float* P[64];
    for (int i = 0; i < n_in && i < 64; ++i) P[i] = (const float*)d_in[i];

    const float *inp, *relr, *rels, *ow, *ob;
    const float *w1[4], *b1a[4], *w2[4], *b2a[4], *mg[4], *mb[4], *bg[4], *bb[4];

    bool insertion = (n_in >= 2 && in_sizes[1] == NEDGE * NNODE);
    if (insertion) {
        inp = P[0]; relr = P[1]; rels = P[2];
        int k = 3;
        for (int m = 0; m < 4; ++m) {
            w1[m] = P[k]; b1a[m] = P[k + 1]; w2[m] = P[k + 2]; b2a[m] = P[k + 3];
            mg[m] = P[k + 4]; mb[m] = P[k + 5]; k += 6;
        }
        for (int m = 0; m < 4; ++m) { bg[m] = P[k]; bb[m] = P[k + 1]; k += 2; }
        ow = P[k]; ob = P[k + 1];
    } else { // jax sorted-key tree order
        inp = P[0];
        int k = 1;
        for (int m = 0; m < 4; ++m) { bb[m] = P[k]; bg[m] = P[k + 1]; k += 2; }
        for (int m = 0; m < 4; ++m) {
            mb[m] = P[k]; b1a[m] = P[k + 1]; b2a[m] = P[k + 2]; mg[m] = P[k + 3];
            w1[m] = P[k + 4]; w2[m] = P[k + 5]; k += 6;
        }
        ob = P[k]; ow = P[k + 1]; relr = P[k + 2]; rels = P[k + 3];
    }

    // ---- workspace carve ----
    char* ws = (char*)d_ws;
    size_t off = 0;
    auto carve = [&](size_t bytes) -> void* {
        off = (off + 255) & ~(size_t)255;
        void* p = ws + off; off += bytes; return p;
    };
    const int Mn = NBATCH * NNODE;   // 2048
    const int Me = NBATCH * NEDGE;   // 63488
    float*  nodeF   = (float*)carve((size_t)Mn * HDIM * 4);
    float*  edgeF   = (float*)carve((size_t)Me * HDIM * 4);
    __bf16* nodeHb  = (__bf16*)carve((size_t)Mn * HDIM * 2);
    __bf16* nodeHb2 = (__bf16*)carve((size_t)Mn * HDIM * 2);
    __bf16* nodeXb  = (__bf16*)carve((size_t)Mn * HDIM * 2);
    __bf16* edgeHb  = (__bf16*)carve((size_t)Me * HDIM * 2);
    __bf16* edgeXb  = (__bf16*)carve((size_t)Me * HDIM * 2);
    __bf16* pw1  = (__bf16*)carve(8  * 16 * 512 * 2);
    __bf16* pw2a = (__bf16*)carve(16 * 16 * 512 * 2);
    __bf16* pw2b = (__bf16*)carve(8  * 16 * 512 * 2);
    __bf16* pw3a = (__bf16*)carve(8  * 16 * 512 * 2);
    __bf16* pw3b = (__bf16*)carve(8  * 16 * 512 * 2);
    __bf16* pw4a = (__bf16*)carve(24 * 16 * 512 * 2);
    __bf16* pw4b = (__bf16*)carve(8  * 16 * 512 * 2);
    int* recI  = (int*)carve(NEDGE * 4);
    int* sendI = (int*)carve(NEDGE * 4);
    int maxCh = (Me + 511) / 512;
    float* psum  = (float*)carve((size_t)maxCh * HDIM * 4);
    float* psum2 = (float*)carve((size_t)maxCh * HDIM * 4);
    float* meanB = (float*)carve(HDIM * 4);
    float* istdB = (float*)carve(HDIM * 4);

    // ---- setup ----
    k_extract_idx<<<(NEDGE + 63) / 64, 64, 0, stream>>>(relr, recI);
    k_extract_idx<<<(NEDGE + 63) / 64, 64, 0, stream>>>(rels, sendI);
    auto pack = [&](const float* W, __bf16* Pk, int K) {
        int total = (K >> 5) * (HDIM >> 4) * 32;
        k_pack_w<<<(total + 255) / 256, 256, 0, stream>>>(W, Pk, K);
    };
    pack(w2[0], pw1, 256);
    pack(w1[1], pw2a, 512); pack(w2[1], pw2b, 256);
    pack(w1[2], pw3a, 256); pack(w2[2], pw3b, 256);
    pack(w1[3], pw4a, 768); pack(w2[3], pw4b, 256);

    auto run_bn = [&](float* X, int M, const float* g, const float* b, __bf16* Xb) {
        int nCh = (M + 511) / 512;
        k_bn_partial<<<nCh, 256, 0, stream>>>(X, psum, psum2, M, 512);
        k_bn_final<<<1, 256, 0, stream>>>(psum, psum2, meanB, istdB, nCh, M);
        k_bn_apply<<<M, 256, 0, stream>>>(X, meanB, istdB, g, b, Xb, M);
    };

    // ---- stage 1: mlp1 + bn1 (node domain) ----
    k_fc8<<<(Mn * HDIM + 255) / 256, 256, 0, stream>>>(inp, w1[0], b1a[0], nodeHb, Mn);
    k_gemm_bf16<<<Mn / 128, 256, 0, stream>>>(nodeHb, nullptr, nullptr, recI, sendI,
                                              pw1, b2a[0], nodeF, nullptr, Mn, 1, 0);
    run_bn(nodeF, Mn, mg[0], mb[0], nullptr);
    run_bn(nodeF, Mn, bg[0], bb[0], nodeXb);        // node1 (bf16)

    // ---- stage 2: node2edge gather fused + mlp2 + bn2 (edge domain) ----
    k_gemm_bf16<<<Me / 128, 256, 0, stream>>>(nullptr, nodeXb, nullptr, recI, sendI,
                                              pw2a, b1a[1], nullptr, edgeHb, Me, 2, 1);
    k_gemm_bf16<<<Me / 128, 256, 0, stream>>>(edgeHb, nullptr, nullptr, recI, sendI,
                                              pw2b, b2a[1], edgeF, nullptr, Me, 1, 0);
    run_bn(edgeF, Me, mg[1], mb[1], nullptr);
    run_bn(edgeF, Me, bg[1], bb[1], edgeXb);        // x_skip: f32 in edgeF, bf16 in edgeXb

    // ---- stage 3: edge2node + mlp3 + bn3 ----
    k_edge2node<<<Mn, 256, 0, stream>>>(edgeF, recI, nodeHb);
    k_gemm_bf16<<<Mn / 128, 256, 0, stream>>>(nodeHb, nullptr, nullptr, recI, sendI,
                                              pw3a, b1a[2], nullptr, nodeHb2, Mn, 1, 0);
    k_gemm_bf16<<<Mn / 128, 256, 0, stream>>>(nodeHb2, nullptr, nullptr, recI, sendI,
                                              pw3b, b2a[2], nodeF, nullptr, Mn, 1, 0);
    run_bn(nodeF, Mn, mg[2], mb[2], nullptr);
    run_bn(nodeF, Mn, bg[2], bb[2], nodeXb);        // node3 (bf16)

    // ---- stage 4: [recv|send|skip] gather fused + mlp4 + bn4 ----
    k_gemm_bf16<<<Me / 128, 256, 0, stream>>>(nullptr, nodeXb, edgeXb, recI, sendI,
                                              pw4a, b1a[3], nullptr, edgeHb, Me, 3, 2);
    k_gemm_bf16<<<Me / 128, 256, 0, stream>>>(edgeHb, nullptr, nullptr, recI, sendI,
                                              pw4b, b2a[3], edgeF, nullptr, Me, 1, 0);
    run_bn(edgeF, Me, mg[3], mb[3], nullptr);
    run_bn(edgeF, Me, bg[3], bb[3], nullptr);

    // ---- output projection + softmax ----
    k_out_softmax<<<(Me + 7) / 8, 256, 0, stream>>>(edgeF, ow, ob, (float*)d_out, Me);
}
